// Policy_35562329211083
// MI455X (gfx1250) — matrix operand order repaired and checked by reading, not compiled, not run on hardware
//
#include <hip/hip_runtime.h>
#include <hip/hip_bf16.h>
#include <math.h>

// ---------------------------------------------------------------------------
// MI455X (gfx1250) fused capsule-routing policy network.
//   K1: WMMA GEMMs (x = relu(inp@W1^T); ac = x@Wa^T; sc = x@Wv^T)
//   K2: Heinsen routing 32 -> 8   (thread = (row, j), shfl_xor softmax over 8)
//   K3: Heinsen routing 8 -> 128 + argmax/sampling tail (wave per row)
// Intermediates live in d_ws (~12.5MB) -> fully L2-resident (192MB L2).
// ---------------------------------------------------------------------------

#define HID 128
#define N1  32
#define N2  8
#define N3  128
#define EPSV 1e-5f
#define XS  144          // padded LDS stride (halves) for x tiles: 288B rows, 16B-aligned sublocks

typedef __attribute__((ext_vector_type(16))) _Float16     v16h;
typedef __attribute__((ext_vector_type(8)))  float        v8f;
typedef __attribute__((ext_vector_type(4)))  unsigned int v4u;

union F16Frag { v16h h; v4u u[2]; };

// ------------------------------ Kernel 1 -----------------------------------
// grid = B/128 blocks x 256 threads (8 waves x 16 rows).
__global__ void __launch_bounds__(256)
mlp_wmma_kernel(const float* __restrict__ inp, const float* __restrict__ W1,
                const float* __restrict__ b1,  const float* __restrict__ Wa,
                const float* __restrict__ ba,  const float* __restrict__ Wv,
                const float* __restrict__ bv,  float* __restrict__ ac,
                float* __restrict__ sc)
{
  extern __shared__ char smem[];
  _Float16* sWa = (_Float16*)smem;          // 128x128 f16 (32KB)
  _Float16* sWv = sWa + HID * HID;          // 32x128 f16 (8KB)
  _Float16* sX  = sWv + N1 * HID;           // 8 waves x 16 x XS f16 (36KB)

  const int tid  = threadIdx.x;
  const int wave = tid >> 5, lane = tid & 31;

  __builtin_prefetch(Wa + tid * 64, 0, 1);  // global_prefetch_b8 warm-up of Wa

  // Stage weights into LDS as f16 (row-major, stride 128 halves = 256B).
  for (int i = tid; i < HID * HID; i += 256) sWa[i] = (_Float16)Wa[i];
  for (int i = tid; i < N1 * HID;  i += 256) sWv[i] = (_Float16)Wv[i];

  const int rowBase = blockIdx.x * 128 + wave * 16;
  _Float16* xt = sX + wave * (16 * XS);

  // x = relu(inp @ W1^T + b1): each lane computes 4 hidden cols for 16 rows.
  {
    const int h0 = lane * 4;
    float w[4][4], bb[4];
    #pragma unroll
    for (int c = 0; c < 4; ++c) {
      bb[c] = b1[h0 + c];
      #pragma unroll
      for (int d = 0; d < 4; ++d) w[c][d] = W1[(h0 + c) * 4 + d];
    }
    const float4* ip = (const float4*)inp;
    #pragma unroll
    for (int m = 0; m < 16; ++m) {
      float4 iv = ip[rowBase + m];
      #pragma unroll
      for (int c = 0; c < 4; ++c) {
        float v = fmaf(iv.x, w[c][0], fmaf(iv.y, w[c][1],
                  fmaf(iv.z, w[c][2], fmaf(iv.w, w[c][3], bb[c]))));
        xt[m * XS + h0 + c] = (_Float16)fmaxf(v, 0.f);
      }
    }
  }
  __syncthreads();

  const int M = lane & 15, hi = lane >> 4;

  // ac = x @ Wa^T + ba : 8 N-tiles x 4 K-chunks of v_wmma_f32_16x16x32_f16
  #pragma unroll
  for (int nt = 0; nt < 8; ++nt) {
    v8f acc = {};
    #pragma unroll
    for (int kc = 0; kc < 4; ++kc) {
      F16Frag A, Bf;
      // A 16x32 f16: lane<16 holds M=lane, K = {kc*32+0..7, kc*32+16..23}; lane>=16: +8
      A.u[0] = *(const v4u*)(xt + M * XS + kc * 32 + hi * 8);
      A.u[1] = *(const v4u*)(xt + M * XS + kc * 32 + 16 + hi * 8);
      // B 32x16 f16: lane%16 = N (= output row of Wa), K = {kc*32+0..7, kc*32+16..23}; lane>=16: +8
      const _Float16* bp = sWa + (nt * 16 + M) * HID + kc * 32 + hi * 8;
      Bf.u[0] = *(const v4u*)(bp);
      Bf.u[1] = *(const v4u*)(bp + 16);
      acc = __builtin_amdgcn_wmma_f32_16x16x32_f16(false, A.h, false, Bf.h,
                                                   (short)0, acc, false, false);
    }
    const int col = nt * 16 + M;
    const float bav = ba[col];
    #pragma unroll
    for (int r = 0; r < 8; ++r)
      ac[(size_t)(rowBase + r + 8 * hi) * HID + col] = acc[r] + bav;
  }

  // sc = x @ Wv^T + bv : 2 N-tiles x 4 K-chunks
  #pragma unroll
  for (int nt = 0; nt < 2; ++nt) {
    v8f acc = {};
    #pragma unroll
    for (int kc = 0; kc < 4; ++kc) {
      F16Frag A, Bf;
      A.u[0] = *(const v4u*)(xt + M * XS + kc * 32 + hi * 8);
      A.u[1] = *(const v4u*)(xt + M * XS + kc * 32 + 16 + hi * 8);
      const _Float16* bp = sWv + (nt * 16 + M) * HID + kc * 32 + hi * 8;
      Bf.u[0] = *(const v4u*)(bp);
      Bf.u[1] = *(const v4u*)(bp + 16);
      acc = __builtin_amdgcn_wmma_f32_16x16x32_f16(false, A.h, false, Bf.h,
                                                   (short)0, acc, false, false);
    }
    const int col = nt * 16 + M;
    const float bvv = bv[col];
    #pragma unroll
    for (int r = 0; r < 8; ++r)
      sc[(size_t)(rowBase + r + 8 * hi) * N1 + col] = acc[r] + bvv;
  }
}

// ------------------------------ Kernel 2 -----------------------------------
// Heinsen routing 32 -> 8.  thread = (row, j): 32 rows/block, j = tid&7.
__global__ void __launch_bounds__(256)
routing1_kernel(const float* __restrict__ ac, const float* __restrict__ sc,
                const float* __restrict__ W,  const float* __restrict__ Bb,
                const float* __restrict__ bu, const float* __restrict__ bi,
                float* __restrict__ a1, float* __restrict__ mu1)
{
  __shared__ float smu[32 * HID];               // pose rows for this block
  const int tid = threadIdx.x;
  const int rl  = tid >> 3;                     // row-in-block 0..31
  const int j   = tid & 7;
  const int row = blockIdx.x * 32 + rl;

  #pragma unroll
  for (int k = 0; k < 16; ++k)
    smu[rl * HID + j * 16 + k] = ac[(size_t)row * HID + j * 16 + k];
  __syncthreads();
  const float* mu = smu + rl * HID;             // [i*4 + c]

  float Wj[16], Bj[4];
  #pragma unroll
  for (int t = 0; t < 16; ++t) Wj[t] = W[j * 16 + t];   // shared W: [1,8,4,4]
  #pragma unroll
  for (int d = 0; d < 4; ++d) Bj[d] = Bb[j * 4 + d];
  const float buj = bu[j], bij = bi[j];

  float fa[N1]; float fasum = 0.f;
  #pragma unroll
  for (int i = 0; i < N1; ++i) {
    float f = 1.f / (1.f + __expf(-sc[(size_t)row * N1 + i]));
    fa[i] = f; fasum += f;
  }

  auto vcomp = [&](int i, float* v) {
    float m0 = mu[i * 4 + 0], m1 = mu[i * 4 + 1], m2 = mu[i * 4 + 2], m3 = mu[i * 4 + 3];
    #pragma unroll
    for (int d = 0; d < 4; ++d)
      v[d] = Bj[d] + m0 * Wj[d] + m1 * Wj[4 + d] + m2 * Wj[8 + d] + m3 * Wj[12 + d];
  };

  float mo[4], sg[4], aout;
  { // iteration 0: R = 1/8
    float Dsum = fasum * 0.125f;
    aout = buj * Dsum - bij * (fasum - Dsum);
    float inv = 1.f / (Dsum + EPSV);
    float macc[4] = {0, 0, 0, 0};
    #pragma unroll
    for (int i = 0; i < N1; ++i) {
      float v[4]; vcomp(i, v);
      float du0 = fa[i] * 0.125f;
      #pragma unroll
      for (int d = 0; d < 4; ++d) macc[d] += du0 * v[d];
    }
    #pragma unroll
    for (int d = 0; d < 4; ++d) mo[d] = macc[d] * inv;
    float sacc[4] = {0, 0, 0, 0};
    #pragma unroll
    for (int i = 0; i < N1; ++i) {
      float v[4]; vcomp(i, v);
      float du0 = fa[i] * 0.125f;
      #pragma unroll
      for (int d = 0; d < 4; ++d) { float df = v[d] - mo[d]; sacc[d] += du0 * df * df; }
    }
    #pragma unroll
    for (int d = 0; d < 4; ++d) sg[d] = sacc[d] * inv + EPSV;
  }

  float du[N1];
  for (int it = 1; it < 3; ++it) {
    float ls = (aout >= 0.f) ? -__logf(1.f + __expf(-aout))
                             : (aout - __logf(1.f + __expf(aout)));
    float hs[4], lc[4];
    #pragma unroll
    for (int d = 0; d < 4; ++d) { hs[d] = -0.5f / sg[d]; lc[d] = -0.5f * __logf(6.28318530718f * sg[d]); }
    float Dsum = 0.f, macc[4] = {0, 0, 0, 0};
    #pragma unroll
    for (int i = 0; i < N1; ++i) {
      float v[4]; vcomp(i, v);
      float lp = 0.f;
      #pragma unroll
      for (int d = 0; d < 4; ++d) { float df = v[d] - mo[d]; lp += hs[d] * df * df + lc[d]; }
      float t = ls + lp;
      float mx = t;
      mx = fmaxf(mx, __shfl_xor(mx, 1, 8));
      mx = fmaxf(mx, __shfl_xor(mx, 2, 8));
      mx = fmaxf(mx, __shfl_xor(mx, 4, 8));
      float e = __expf(t - mx);
      float ss = e;
      ss += __shfl_xor(ss, 1, 8);
      ss += __shfl_xor(ss, 2, 8);
      ss += __shfl_xor(ss, 4, 8);
      float d_ = fa[i] * (e / ss);
      du[i] = d_; Dsum += d_;
      #pragma unroll
      for (int d = 0; d < 4; ++d) macc[d] += d_ * v[d];
    }
    aout = buj * Dsum - bij * (fasum - Dsum);
    float inv = 1.f / (Dsum + EPSV);
    #pragma unroll
    for (int d = 0; d < 4; ++d) mo[d] = macc[d] * inv;
    float sacc[4] = {0, 0, 0, 0};
    #pragma unroll
    for (int i = 0; i < N1; ++i) {
      float v[4]; vcomp(i, v);
      #pragma unroll
      for (int d = 0; d < 4; ++d) { float df = v[d] - mo[d]; sacc[d] += du[i] * df * df; }
    }
    #pragma unroll
    for (int d = 0; d < 4; ++d) sg[d] = sacc[d] * inv + EPSV;
  }

  a1[(size_t)row * N2 + j] = aout;
  #pragma unroll
  for (int d = 0; d < 4; ++d) mu1[((size_t)row * N2 + j) * 4 + d] = mo[d];
}

// ------------------------------ Kernel 3 -----------------------------------
__device__ inline float urand(unsigned s) {
  s = s * 747796405u + 2891336453u;
  unsigned w = ((s >> ((s >> 28u) + 4u)) ^ s) * 277803737u;
  w = (w >> 22u) ^ w;
  return ((float)w + 0.5f) * (1.0f / 4294967296.0f);
}
__device__ inline float nrand(unsigned s) {
  float u1 = urand(s * 2654435761u + 1u);
  float u2 = urand(s * 0x9E3779B9u + 7u);
  return sqrtf(-2.f * __logf(u1)) * __cosf(6.28318530718f * u2);
}

// Heinsen routing 8 -> 128 + tail. wave per row, 4 output caps per lane.
__global__ void __launch_bounds__(256)
routing2_final_kernel(const float* __restrict__ a1, const float* __restrict__ mu1,
                      const float* __restrict__ W,  const float* __restrict__ Bb,
                      const float* __restrict__ bu, const float* __restrict__ bi,
                      const float* __restrict__ inp, const float* __restrict__ subw,
                      float* __restrict__ out, int Btot)
{
  extern __shared__ float sW[];                // [8][128][16] = 64KB
  const int tid = threadIdx.x, wave = tid >> 5, lane = tid & 31;
  for (int t = tid; t < N2 * N3 * 16; t += 256) sW[t] = W[t];
  __syncthreads();

  const int row = blockIdx.x * 8 + wave;

  float fa[N2], mi[N2][4]; float fasum = 0.f;
  #pragma unroll
  for (int i = 0; i < N2; ++i) {
    float a = a1[(size_t)row * N2 + i];
    fa[i] = 1.f / (1.f + __expf(-a)); fasum += fa[i];
    #pragma unroll
    for (int c = 0; c < 4; ++c) mi[i][c] = mu1[((size_t)row * N2 + i) * 4 + c];
  }

  float Bj[4][4], buq[4], biq[4];
  #pragma unroll
  for (int q = 0; q < 4; ++q) {
    int j = q * 32 + lane;
    buq[q] = bu[j]; biq[q] = bi[j];
    #pragma unroll
    for (int d = 0; d < 4; ++d) Bj[q][d] = Bb[j * 4 + d];
  }

  auto vcomp = [&](int i, int q, float* v) {
    int j = q * 32 + lane;
    const float* w = sW + (i * N3 + j) * 16;
    #pragma unroll
    for (int d = 0; d < 4; ++d)
      v[d] = Bj[q][d] + mi[i][0] * w[d] + mi[i][1] * w[4 + d]
                      + mi[i][2] * w[8 + d] + mi[i][3] * w[12 + d];
  };

  float mo[4][4], sg[4][4], aout[4];
  { // iteration 0: R = 1/128
    float Dsum = fasum * (1.f / 128.f);
    float inv = 1.f / (Dsum + EPSV);
    #pragma unroll
    for (int q = 0; q < 4; ++q) aout[q] = buq[q] * Dsum - biq[q] * (fasum - Dsum);
    float macc[4][4] = {};
    #pragma unroll
    for (int i = 0; i < N2; ++i) {
      float du0 = fa[i] * (1.f / 128.f);
      #pragma unroll
      for (int q = 0; q < 4; ++q) {
        float v[4]; vcomp(i, q, v);
        #pragma unroll
        for (int d = 0; d < 4; ++d) macc[q][d] += du0 * v[d];
      }
    }
    #pragma unroll
    for (int q = 0; q < 4; ++q)
      #pragma unroll
      for (int d = 0; d < 4; ++d) mo[q][d] = macc[q][d] * inv;
    float sacc[4][4] = {};
    #pragma unroll
    for (int i = 0; i < N2; ++i) {
      float du0 = fa[i] * (1.f / 128.f);
      #pragma unroll
      for (int q = 0; q < 4; ++q) {
        float v[4]; vcomp(i, q, v);
        #pragma unroll
        for (int d = 0; d < 4; ++d) { float df = v[d] - mo[q][d]; sacc[q][d] += du0 * df * df; }
      }
    }
    #pragma unroll
    for (int q = 0; q < 4; ++q)
      #pragma unroll
      for (int d = 0; d < 4; ++d) sg[q][d] = sacc[q][d] * inv + EPSV;
  }

  float du[N2][4];
  for (int it = 1; it < 3; ++it) {
    float ls[4], hs[4][4], lc[4][4];
    #pragma unroll
    for (int q = 0; q < 4; ++q) {
      float a = aout[q];
      ls[q] = (a >= 0.f) ? -__logf(1.f + __expf(-a)) : (a - __logf(1.f + __expf(a)));
      #pragma unroll
      for (int d = 0; d < 4; ++d) {
        hs[q][d] = -0.5f / sg[q][d];
        lc[q][d] = -0.5f * __logf(6.28318530718f * sg[q][d]);
      }
    }
    float Dsum[4] = {0, 0, 0, 0}, macc[4][4] = {};
    #pragma unroll
    for (int i = 0; i < N2; ++i) {
      float vq[4][4], t[4];
      float mloc = -3.402823e38f;
      #pragma unroll
      for (int q = 0; q < 4; ++q) {
        vcomp(i, q, vq[q]);
        float lp = 0.f;
        #pragma unroll
        for (int d = 0; d < 4; ++d) { float df = vq[q][d] - mo[q][d]; lp += hs[q][d] * df * df + lc[q][d]; }
        t[q] = ls[q] + lp;
        mloc = fmaxf(mloc, t[q]);
      }
      #pragma unroll
      for (int off = 16; off >= 1; off >>= 1) mloc = fmaxf(mloc, __shfl_xor(mloc, off, 32));
      float e[4], ssum = 0.f;
      #pragma unroll
      for (int q = 0; q < 4; ++q) { e[q] = __expf(t[q] - mloc); ssum += e[q]; }
      #pragma unroll
      for (int off = 16; off >= 1; off >>= 1) ssum += __shfl_xor(ssum, off, 32);
      float rs = 1.f / ssum;
      #pragma unroll
      for (int q = 0; q < 4; ++q) {
        float d_ = fa[i] * e[q] * rs;
        du[i][q] = d_; Dsum[q] += d_;
        #pragma unroll
        for (int d = 0; d < 4; ++d) macc[q][d] += d_ * vq[q][d];
      }
    }
    #pragma unroll
    for (int q = 0; q < 4; ++q) {
      aout[q] = buq[q] * Dsum[q] - biq[q] * (fasum - Dsum[q]);
      float inv = 1.f / (Dsum[q] + EPSV);
      #pragma unroll
      for (int d = 0; d < 4; ++d) mo[q][d] = macc[q][d] * inv;
    }
    float sacc[4][4] = {};
    #pragma unroll
    for (int i = 0; i < N2; ++i) {
      #pragma unroll
      for (int q = 0; q < 4; ++q) {
        float v[4]; vcomp(i, q, v);
        #pragma unroll
        for (int d = 0; d < 4; ++d) { float df = v[d] - mo[q][d]; sacc[q][d] += du[i][q] * df * df; }
      }
    }
    #pragma unroll
    for (int q = 0; q < 4; ++q) {
      float inv = 1.f / (Dsum[q] + EPSV);
      #pragma unroll
      for (int d = 0; d < 4; ++d) sg[q][d] = sacc[q][d] * inv + EPSV;
    }
  }

  // ---- tail: value = max a_out, argmax pose -> softmax -> sampled head ----
  float vmax = aout[0]; int qb = 0;
  #pragma unroll
  for (int q = 1; q < 4; ++q) if (aout[q] > vmax) { vmax = aout[q]; qb = q; }
  float vred = vmax; int jred = qb * 32 + lane;
  #pragma unroll
  for (int off = 16; off >= 1; off >>= 1) {
    float vo = __shfl_xor(vred, off, 32);
    int   jo = __shfl_xor(jred, off, 32);
    if (vo > vred || (vo == vred && jo < jred)) { vred = vo; jred = jo; }
  }
  const int srcl = jred & 31, qs = jred >> 5;   // uniform across the wave
  float sel[4];
  #pragma unroll
  for (int d = 0; d < 4; ++d) {
    float mqd = (qs == 0) ? mo[0][d] : (qs == 1) ? mo[1][d] : (qs == 2) ? mo[2][d] : mo[3][d];
    sel[d] = __shfl(mqd, srcl, 32);
  }
  float smax = fmaxf(fmaxf(sel[0], sel[1]), fmaxf(sel[2], sel[3]));
  float p[4], psum = 0.f;
  #pragma unroll
  for (int d = 0; d < 4; ++d) { p[d] = __expf(sel[d] - smax); psum += p[d]; }
  #pragma unroll
  for (int d = 0; d < 4; ++d) p[d] /= psum;

  unsigned seed = (unsigned)row * 2654435761u + 0xC0FFEEu;
  float u = urand(seed);
  int sub = 0; float cacc = p[0];
  #pragma unroll
  for (int d = 0; d < 3; ++d) { if (u > cacc) { ++sub; cacc += p[sub]; } }

  const float4 iv = ((const float4*)inp)[row];
  const float inr[4] = {iv.x, iv.y, iv.z, iv.w};
  float fin[2] = {0.f, 0.f};
  #pragma unroll
  for (int o = 0; o < 2; ++o)
    #pragma unroll
    for (int c = 0; c < 4; ++c) {
      float e = nrand(seed ^ (0x55u + o * 4u + c));
      float w = subw[o * 4 + c];
      w = (sub == 0) ? w * e : w / e;
      w = fminf(1.f, fmaxf(-1.f, w));
      fin[o] += inr[c] * w;
    }
  float fm = fmaxf(fin[0], fin[1]);
  float e0 = __expf(fin[0] - fm), e1 = __expf(fin[1] - fm);
  float p0 = e0 / (e0 + e1), p1 = 1.f - p0;
  float u2 = urand(seed ^ 0xBADC0DEu);
  int action = (u2 > p0) ? 1 : 0;
  float lprob = __logf(((action == 0) ? p0 : p1) + 1e-20f);

  if (lane == 0) {
    out[row]             = (float)action;
    out[Btot + row]      = lprob;
    out[2 * Btot + row]  = vred;
  }
}

// ------------------------------ launcher -----------------------------------
extern "C" void kernel_launch(void* const* d_in, const int* in_sizes, int n_in,
                              void* d_out, int out_size, void* d_ws, size_t ws_size,
                              hipStream_t stream) {
  const float* inp  = (const float*)d_in[0];
  const float* W1   = (const float*)d_in[1];
  const float* b1   = (const float*)d_in[2];
  const float* Wa   = (const float*)d_in[3];
  const float* ba   = (const float*)d_in[4];
  const float* Wv   = (const float*)d_in[5];
  const float* bv   = (const float*)d_in[6];
  const float* R1W  = (const float*)d_in[7];
  const float* R1B  = (const float*)d_in[8];
  const float* R1bu = (const float*)d_in[9];
  const float* R1bi = (const float*)d_in[10];
  const float* R2W  = (const float*)d_in[11];
  const float* R2B  = (const float*)d_in[12];
  const float* R2bu = (const float*)d_in[13];
  const float* R2bi = (const float*)d_in[14];
  const float* subw = (const float*)d_in[15];

  const int Btot = in_sizes[0] / 4;          // 16384

  float* ws  = (float*)d_ws;
  float* ac  = ws;                           // [B,128]
  float* sc  = ac + (size_t)Btot * HID;      // [B,32]
  float* a1  = sc + (size_t)Btot * N1;       // [B,8]
  float* mu1 = a1 + (size_t)Btot * N2;       // [B,8,4]
  float* out = (float*)d_out;                // action | log_prob | value

  const size_t smem1 = (size_t)(HID * HID + N1 * HID) * sizeof(_Float16)
                     + (size_t)8 * 16 * XS * sizeof(_Float16);           // ~76KB
  mlp_wmma_kernel<<<Btot / 128, 256, smem1, stream>>>(inp, W1, b1, Wa, ba, Wv, bv, ac, sc);

  routing1_kernel<<<Btot / 32, 256, 0, stream>>>(ac, sc, R1W, R1B, R1bu, R1bi, a1, mu1);

  const size_t smem3 = (size_t)N2 * N3 * 16 * sizeof(float);             // 64KB
  routing2_final_kernel<<<Btot / 8, 256, smem3, stream>>>(a1, mu1, R2W, R2B, R2bu, R2bi,
                                                          inp, subw, out, Btot);
}
